// SchNetAuTopology_81398220194556
// MI455X (gfx1250) — matrix operand order repaired and checked by reading, not compiled, not run on hardware
//
#include <hip/hip_runtime.h>
#include <hip/hip_bf16.h>

// ---------------------------------------------------------------------------
// SchNet-like GNN forward for MI455X (gfx1250, wave32, WMMA f16->f32)
// ---------------------------------------------------------------------------

typedef __attribute__((ext_vector_type(16))) _Float16 v16h;
typedef __attribute__((ext_vector_type(8)))  _Float16 v8h;
typedef __attribute__((ext_vector_type(8)))  float    v8f;
typedef __attribute__((ext_vector_type(4)))  int      v4i;

#define DEVI __device__ __forceinline__

// Compiler barrier: stops LICM from hoisting the (loop-invariant) LDS
// B-fragment loads out of the tile loop, which would spill ~256 VGPRs of
// fragments to scratch.  Re-loading from LDS per tile is far cheaper.
#define NO_HOIST() asm volatile("" ::: "memory")

constexpr int kAtoms = 50000;
constexpr int kEdges = 1500000;
constexpr int kMols  = 1000;
constexpr int kNB = 128, kNF = 128, kNG = 50, kNH = 64;
constexpr int kTilesA = kAtoms / 16;   // 3125
constexpr int kTilesE = kEdges / 16;   // 93750
constexpr float kInvWidth = 49.0f / 5.0f;   // 1/width, width = CUTOFF/(NG-1)

// f16 transposed-weight buffer layout (offsets in halves)
constexpr int OWIN  = 0;                       // WinT  [l][n=128][k=128]
constexpr int OWF2  = 49152;                   // Wf2T  [l][n=128][k=128]
constexpr int OWOUT = 98304;                   // WoutT [l][n=128][k=128]
constexpr int OWF1  = 147456;                  // Wf1T  [l][n=128][k=64] (k>=50 zero-padded)
constexpr int OWR1  = 172032;                  // Wr1T  [kk][n=64][k=128]
constexpr int W16_TOTAL = 196608;

// workspace byte offsets
constexpr size_t WS_R    = 0;                  // float[50000*128]
constexpr size_t WS_AGG  = 25600000;           // float[50000*128]
constexpr size_t WS_H16  = 51200000;           // half [50000*128]
constexpr size_t WS_MOLE = 64000000;           // float[3*1000]
constexpr size_t WS_W16  = 64012032;           // half [W16_TOTAL]

DEVI v8f v8f_zero() {
    v8f z;
#pragma unroll
    for (int t = 0; t < 8; ++t) z[t] = 0.0f;
    return z;
}

DEVI v8f wmma32(v16h a, v16h b, v8f c) {
    // D = A(16x32 f16) * B(32x16 f16) + C(16x16 f32)
    return __builtin_amdgcn_wmma_f32_16x16x32_f16(
        /*neg_a=*/false, a, /*neg_b=*/false, b,
        /*c_mod=*/(short)0, c, /*reuse_a=*/false, /*reuse_b=*/false);
}

// shifted softplus: log(1+exp(x)) - log(2), numerically stable
DEVI float sspf(float x) {
    float ax = fabsf(x);
    return fmaxf(x, 0.0f) + __logf(1.0f + __expf(-ax)) - 0.69314718056f;
}

// Build A-fragment from an f32 row (atom features), optional ssp activation.
// Halves h=0..7 -> k = koff+h ; h=8..15 -> k = koff+8+(h-8)
DEVI v16h a_from_f32(const float* row, int koff, bool act) {
    v8f lo = *(const v8f*)(row + koff);
    v8f hi = *(const v8f*)(row + koff + 8);
    v16h a;
#pragma unroll
    for (int t = 0; t < 8; ++t) {
        float x = lo[t], y = hi[t];
        if (act) { x = sspf(x); y = sspf(y); }
        a[t]     = (_Float16)x;
        a[t + 8] = (_Float16)y;
    }
    return a;
}

// ---------------------------------------------------------------------------
// Global -> LDS staging.  Prefer the CDNA5 async-LDS path (ASYNCcnt) when the
// toolchain exposes it; otherwise fall back to plain per-thread copies.
// Caller must __syncthreads() afterwards (async completion is waited here).
// ---------------------------------------------------------------------------
#if defined(__AMDGCN__) && __has_builtin(__builtin_amdgcn_global_load_async_to_lds_b128)
#define HAVE_ASYNC_LDS 1
#else
#define HAVE_ASYNC_LDS 0
#endif

DEVI void async_wait0() {
#if defined(__AMDGCN__) && __has_builtin(__builtin_amdgcn_s_wait_asynccnt)
    __builtin_amdgcn_s_wait_asynccnt(0);
#else
    asm volatile("s_wait_asynccnt 0" ::: "memory");
#endif
}

DEVI void stage_lds(_Float16* dstLds, const _Float16* srcGlobal, int halves) {
#if HAVE_ASYNC_LDS
    for (int t = threadIdx.x * 8; t < halves; t += blockDim.x * 8) {
        __builtin_amdgcn_global_load_async_to_lds_b128(
            (__attribute__((address_space(1))) v4i*)(srcGlobal + t),
            (__attribute__((address_space(3))) v4i*)(dstLds + t),
            /*offset=*/0, /*cpol=*/0);
    }
    async_wait0();
#else
    for (int t = threadIdx.x; t < halves; t += blockDim.x) dstLds[t] = srcGlobal[t];
#endif
}

// ---------------------------------------------------------------------------
// init: r = embed[z], agg = 0, mol_e = 0
// ---------------------------------------------------------------------------
__global__ void init_kernel(const int* __restrict__ z, const float* __restrict__ embed,
                            float* __restrict__ r, float* __restrict__ agg,
                            float* __restrict__ mol_e) {
    size_t idx = (size_t)blockIdx.x * blockDim.x + threadIdx.x;
    if (idx < (size_t)kAtoms * kNB) {
        int atom = (int)(idx >> 7), f = (int)(idx & 127);
        r[idx]   = embed[z[atom] * kNB + f];
        agg[idx] = 0.0f;
    }
    if (idx < 3 * kMols) mol_e[idx] = 0.0f;
}

// ---------------------------------------------------------------------------
// convert all GEMM weights to f16, transposed [n][k]
// ---------------------------------------------------------------------------
__global__ void convert_kernel(const float* __restrict__ Win, const float* __restrict__ Wf1,
                               const float* __restrict__ Wf2, const float* __restrict__ Wout,
                               const float* __restrict__ Wr1, _Float16* __restrict__ w16) {
    int idx = blockIdx.x * blockDim.x + threadIdx.x;
    if (idx >= W16_TOTAL) return;
    float val;
    if (idx < OWF2) {                    // WinT
        int t = idx, l = t / 16384, rem = t % 16384, n = rem / 128, k = rem % 128;
        val = Win[l * 16384 + k * 128 + n];
    } else if (idx < OWOUT) {            // Wf2T
        int t = idx - OWF2, l = t / 16384, rem = t % 16384, n = rem / 128, k = rem % 128;
        val = Wf2[l * 16384 + k * 128 + n];
    } else if (idx < OWF1) {             // WoutT
        int t = idx - OWOUT, l = t / 16384, rem = t % 16384, n = rem / 128, k = rem % 128;
        val = Wout[l * 16384 + k * 128 + n];
    } else if (idx < OWR1) {             // Wf1T (pad K 50->64)
        int t = idx - OWF1, l = t / 8192, rem = t % 8192, n = rem / 64, k = rem % 64;
        val = (k < kNG) ? Wf1[l * kNG * 128 + k * 128 + n] : 0.0f;
    } else {                             // Wr1T [kk][h][b]
        int t = idx - OWR1, kk = t / 8192, rem = t % 8192, h = rem / 128, b = rem % 128;
        val = Wr1[kk * 128 * 64 + b * 64 + h];
    }
    w16[idx] = (_Float16)val;
}

// ---------------------------------------------------------------------------
// h = f16(r @ Win[l] + b_in[l]) : 16-atom tiles, grid-stride per wave
// ---------------------------------------------------------------------------
__global__ void h_kernel(const float* __restrict__ r, const _Float16* __restrict__ w16,
                         const float* __restrict__ b_in, _Float16* __restrict__ h16, int layer) {
    __shared__ _Float16 sW[128 * 128];
    stage_lds(sW, w16 + OWIN + layer * 16384, 16384);
    __syncthreads();

    int wave = threadIdx.x >> 5, lane = threadIdx.x & 31;
    int mA = lane & 15;
    int kbase = (lane < 16) ? 0 : 16;
    int kbB   = (lane < 16) ? 0 : 16;
    int mbase = (lane < 16) ? 0 : 8;

    for (int tile = blockIdx.x * 4 + wave; tile < kTilesA; tile += gridDim.x * 4) {
        NO_HOIST();
        const float* arow = r + (size_t)(tile * 16 + mA) * 128;
        v8f c[8];
#pragma unroll
        for (int nt = 0; nt < 8; ++nt) c[nt] = v8f_zero();

        for (int ks = 0; ks < 4; ++ks) {
            v16h a = a_from_f32(arow, ks * 32 + kbase, false);
#pragma unroll
            for (int nt = 0; nt < 8; ++nt) {
                v16h b = *(const v16h*)(&sW[(nt * 16 + mA) * 128 + ks * 32 + kbB]);
                c[nt] = wmma32(a, b, c[nt]);
            }
        }
#pragma unroll
        for (int nt = 0; nt < 8; ++nt) {
            int n = nt * 16 + mA;
            float bi = b_in[layer * kNF + n];
#pragma unroll
            for (int rr = 0; rr < 8; ++rr) {
                size_t atom = (size_t)(tile * 16 + rr + mbase);
                h16[atom * 128 + n] = (_Float16)(c[nt][rr] + bi);
            }
        }
    }
}

// ---------------------------------------------------------------------------
// edge kernel: smear -> WMMA(50x128) -> ssp -> WMMA(128x128) -> *h[j] -> atomic agg[i]
// 16-edge tiles, grid-stride per wave; Wf1T/Wf2T staged in LDS once per workgroup
// ---------------------------------------------------------------------------
__global__ void edge_kernel(const float* __restrict__ xyz, const int* __restrict__ nbr,
                            const _Float16* __restrict__ w16, const float* __restrict__ bf1,
                            const float* __restrict__ bf2, const _Float16* __restrict__ h16,
                            float* __restrict__ agg, int layer) {
    __shared__ _Float16 sWf1[128 * 64];    // 16 KB
    __shared__ _Float16 sWf2[128 * 128];   // 32 KB
    __shared__ _Float16 sT[4][16 * 128];   // 4 KB per wave
    stage_lds(sWf1, w16 + OWF1 + layer * (128 * 64), 128 * 64);
    stage_lds(sWf2, w16 + OWF2 + layer * (128 * 128), 128 * 128);
    __syncthreads();

    int wave = threadIdx.x >> 5, lane = threadIdx.x & 31;
    int mA = lane & 15;
    int kbase = (lane < 16) ? 0 : 16;
    int kbB   = (lane < 16) ? 0 : 16;
    int mbase = (lane < 16) ? 0 : 8;

    for (int tile = blockIdx.x * 4 + wave; tile < kTilesE; tile += gridDim.x * 4) {
        NO_HOIST();
        // distance for "my" A-row edge (scaled by 1/width)
        int e = tile * 16 + mA;
        int i0 = nbr[2 * e], j0 = nbr[2 * e + 1];
        float dx = xyz[3 * i0 + 0] - xyz[3 * j0 + 0];
        float dy = xyz[3 * i0 + 1] - xyz[3 * j0 + 1];
        float dz = xyz[3 * i0 + 2] - xyz[3 * j0 + 2];
        float ds = sqrtf(dx * dx + dy * dy + dz * dz) * kInvWidth;

        // per-C-row edge endpoints (for gather/scatter); prefetch h rows
        int iiR[8], jjR[8];
#pragma unroll
        for (int rr = 0; rr < 8; ++rr) {
            int ee = tile * 16 + rr + mbase;
            iiR[rr] = nbr[2 * ee];
            jjR[rr] = nbr[2 * ee + 1];
            __builtin_prefetch(&h16[(size_t)jjR[rr] * 128], 0, 1);
        }

        // ---- GEMM1: smear(16x64, padded) @ Wf1(64x128) ----
        v8f c[8];
#pragma unroll
        for (int nt = 0; nt < 8; ++nt) c[nt] = v8f_zero();

#pragma unroll
        for (int ks = 0; ks < 2; ++ks) {
            v16h a;
#pragma unroll
            for (int h = 0; h < 16; ++h) {
                int k = ks * 32 + kbase + h;
                float u = ds - (float)k;
                a[h] = (_Float16)((k < kNG) ? __expf(-0.5f * u * u) : 0.0f);
            }
#pragma unroll
            for (int nt = 0; nt < 8; ++nt) {
                v16h b = *(const v16h*)(&sWf1[(nt * 16 + mA) * 64 + ks * 32 + kbB]);
                c[nt] = wmma32(a, b, c[nt]);
            }
        }

        // ssp activation + re-layout (C layout -> A layout) via per-wave LDS tile
#pragma unroll
        for (int nt = 0; nt < 8; ++nt) {
            int n = nt * 16 + mA;
            float b1 = bf1[layer * kNF + n];
#pragma unroll
            for (int rr = 0; rr < 8; ++rr)
                sT[wave][(rr + mbase) * 128 + n] = (_Float16)sspf(c[nt][rr] + b1);
        }
        // same-wave DS ops are in-order: no workgroup barrier needed

        // ---- GEMM2: act(16x128) @ Wf2(128x128) ----
#pragma unroll
        for (int nt = 0; nt < 8; ++nt) c[nt] = v8f_zero();

#pragma unroll
        for (int ks = 0; ks < 4; ++ks) {
            int koff = ks * 32 + kbase;
            v8h lo = *(const v8h*)(&sT[wave][mA * 128 + koff]);
            v8h hi = *(const v8h*)(&sT[wave][mA * 128 + koff + 8]);
            v16h a;
#pragma unroll
            for (int t = 0; t < 8; ++t) { a[t] = lo[t]; a[t + 8] = hi[t]; }
#pragma unroll
            for (int nt = 0; nt < 8; ++nt) {
                v16h b = *(const v16h*)(&sWf2[(nt * 16 + mA) * 128 + ks * 32 + kbB]);
                c[nt] = wmma32(a, b, c[nt]);
            }
        }

        // msg = (filt+bf2) * h[j];  agg[i] += msg   (C layout: lane owns column n)
#pragma unroll
        for (int nt = 0; nt < 8; ++nt) {
            int n = nt * 16 + mA;
            float b2 = bf2[layer * kNF + n];
#pragma unroll
            for (int rr = 0; rr < 8; ++rr) {
                float filt = c[nt][rr] + b2;
                float msg = filt * (float)h16[(size_t)jjR[rr] * 128 + n];
                atomicAdd(&agg[(size_t)iiR[rr] * 128 + n], msg);
            }
        }
    }
}

// ---------------------------------------------------------------------------
// r += ssp(agg) @ Wout[l] + bout[l];  also re-zero agg for the next layer
// ---------------------------------------------------------------------------
__global__ void upd_kernel(float* __restrict__ agg, const _Float16* __restrict__ w16,
                           const float* __restrict__ bout, float* __restrict__ r, int layer) {
    __shared__ _Float16 sW[128 * 128];
    stage_lds(sW, w16 + OWOUT + layer * 16384, 16384);
    __syncthreads();

    int wave = threadIdx.x >> 5, lane = threadIdx.x & 31;
    int mA = lane & 15;
    int kbase = (lane < 16) ? 0 : 16;
    int kbB   = (lane < 16) ? 0 : 16;
    int mbase = (lane < 16) ? 0 : 8;

    for (int tile = blockIdx.x * 4 + wave; tile < kTilesA; tile += gridDim.x * 4) {
        NO_HOIST();
        float* arow = agg + (size_t)(tile * 16 + mA) * 128;
        v8f c[8];
#pragma unroll
        for (int nt = 0; nt < 8; ++nt) c[nt] = v8f_zero();

        for (int ks = 0; ks < 4; ++ks) {
            int koff = ks * 32 + kbase;
            v16h a = a_from_f32(arow, koff, true);           // ssp applied
            // zero the agg elements this lane consumed (each element read exactly once)
            *(v8f*)(arow + koff)      = v8f_zero();
            *(v8f*)(arow + koff + 8) = v8f_zero();
#pragma unroll
            for (int nt = 0; nt < 8; ++nt) {
                v16h b = *(const v16h*)(&sW[(nt * 16 + mA) * 128 + ks * 32 + kbB]);
                c[nt] = wmma32(a, b, c[nt]);
            }
        }
#pragma unroll
        for (int nt = 0; nt < 8; ++nt) {
            int n = nt * 16 + mA;
            float bo = bout[layer * kNB + n];
#pragma unroll
            for (int rr = 0; rr < 8; ++rr) {
                size_t idx = (size_t)(tile * 16 + rr + mbase) * 128 + n;
                r[idx] = r[idx] + c[nt][rr] + bo;
            }
        }
    }
}

// ---------------------------------------------------------------------------
// readout: for kk=0..2: ha = ssp(r @ Wr1[kk] + br1); atom_e = ha @ Wr2[kk] + br2
//          mol_e[kk][mol_idx[atom]] += atom_e
// ---------------------------------------------------------------------------
__global__ void readout_kernel(const float* __restrict__ r, const _Float16* __restrict__ w16,
                               const float* __restrict__ br1, const float* __restrict__ Wr2,
                               const float* __restrict__ br2, const int* __restrict__ mol_idx,
                               float* __restrict__ mol_e) {
    __shared__ _Float16 sW[3 * 64 * 128];   // 48 KB
    stage_lds(sW, w16 + OWR1, 3 * 64 * 128);
    __syncthreads();

    int wave = threadIdx.x >> 5, lane = threadIdx.x & 31;
    int mA = lane & 15;
    int kbase = (lane < 16) ? 0 : 16;
    int kbB   = (lane < 16) ? 0 : 16;
    int mbase = (lane < 16) ? 0 : 8;

    for (int tile = blockIdx.x * 4 + wave; tile < kTilesA; tile += gridDim.x * 4) {
        NO_HOIST();
        const float* arow = r + (size_t)(tile * 16 + mA) * 128;
        v16h aA[4];
#pragma unroll
        for (int ks = 0; ks < 4; ++ks) aA[ks] = a_from_f32(arow, ks * 32 + kbase, false);

        for (int kk = 0; kk < 3; ++kk) {
            NO_HOIST();
            v8f c[4];
#pragma unroll
            for (int nt = 0; nt < 4; ++nt) c[nt] = v8f_zero();
#pragma unroll
            for (int ks = 0; ks < 4; ++ks) {
#pragma unroll
                for (int nt = 0; nt < 4; ++nt) {
                    v16h b = *(const v16h*)(&sW[kk * 8192 + (nt * 16 + mA) * 128 + ks * 32 + kbB]);
                    c[nt] = wmma32(aA[ks], b, c[nt]);
                }
            }
            float s[8];
#pragma unroll
            for (int rr = 0; rr < 8; ++rr) s[rr] = 0.0f;
#pragma unroll
            for (int nt = 0; nt < 4; ++nt) {
                int n = nt * 16 + mA;
                float b1 = br1[kk * kNH + n];
                float w2 = Wr2[kk * kNH + n];
#pragma unroll
                for (int rr = 0; rr < 8; ++rr) s[rr] += sspf(c[nt][rr] + b1) * w2;
            }
            // butterfly sum over the 16 lanes that share each C-row
#pragma unroll
            for (int mask = 1; mask < 16; mask <<= 1) {
#pragma unroll
                for (int rr = 0; rr < 8; ++rr) s[rr] += __shfl_xor(s[rr], mask, 32);
            }
            if ((lane & 15) == 0) {
                float b2 = br2[kk];
#pragma unroll
                for (int rr = 0; rr < 8; ++rr) {
                    int atom = tile * 16 + rr + mbase;
                    atomicAdd(&mol_e[kk * kMols + mol_idx[atom]], s[rr] + b2);
                }
            }
        }
    }
}

// ---------------------------------------------------------------------------
// final 2x2 symmetric eigensolve per molecule
// ---------------------------------------------------------------------------
__global__ void final_kernel(const float* __restrict__ mol_e, float* __restrict__ out) {
    int tid = blockIdx.x * blockDim.x + threadIdx.x;
    if (tid >= kMols) return;
    float d0 = mol_e[tid], d1 = mol_e[kMols + tid], lam = mol_e[2 * kMols + tid];
    float mean = 0.5f * (d0 + d1);
    float diff = d0 - d1;
    float root = sqrtf(0.25f * diff * diff + lam * lam);
    out[tid]         = mean - root;
    out[kMols + tid] = mean + root;
}

// ---------------------------------------------------------------------------
extern "C" void kernel_launch(void* const* d_in, const int* in_sizes, int n_in,
                              void* d_out, int out_size, void* d_ws, size_t ws_size,
                              hipStream_t stream) {
    const int*   z       = (const int*)d_in[0];
    const float* xyz     = (const float*)d_in[1];
    const int*   nbr     = (const int*)d_in[2];
    const int*   mol_idx = (const int*)d_in[3];
    const float* embed   = (const float*)d_in[4];
    const float* Win     = (const float*)d_in[5];
    const float* b_in    = (const float*)d_in[6];
    const float* Wf1     = (const float*)d_in[7];
    const float* bf1     = (const float*)d_in[8];
    const float* Wf2     = (const float*)d_in[9];
    const float* bf2     = (const float*)d_in[10];
    const float* Wout    = (const float*)d_in[11];
    const float* bout    = (const float*)d_in[12];
    const float* Wr1     = (const float*)d_in[13];
    const float* br1     = (const float*)d_in[14];
    const float* Wr2     = (const float*)d_in[15];
    const float* br2     = (const float*)d_in[16];

    char* ws = (char*)d_ws;
    float*     rbuf  = (float*)(ws + WS_R);
    float*     agg   = (float*)(ws + WS_AGG);
    _Float16*  h16   = (_Float16*)(ws + WS_H16);
    float*     mol_e = (float*)(ws + WS_MOLE);
    _Float16*  w16   = (_Float16*)(ws + WS_W16);

    init_kernel<<<(kAtoms * kNB + 255) / 256, 256, 0, stream>>>(z, embed, rbuf, agg, mol_e);
    convert_kernel<<<W16_TOTAL / 256, 256, 0, stream>>>(Win, Wf1, Wf2, Wout, Wr1, w16);

    const int blocksA = (kTilesA + 3) / 4;        // ~1 tile per wave, machine stays full
    const int blocksE = (kTilesE + 15) / 16;      // ~4 tiles per wave, amortize LDS staging
    for (int l = 0; l < 3; ++l) {
        h_kernel   <<<blocksA, 128, 0, stream>>>(rbuf, w16, b_in, h16, l);
        edge_kernel<<<blocksE, 128, 0, stream>>>(xyz, nbr, w16, bf1, bf2, h16, agg, l);
        upd_kernel <<<blocksA, 128, 0, stream>>>(agg, w16, bout, rbuf, l);
    }
    readout_kernel<<<blocksA, 128, 0, stream>>>(rbuf, w16, br1, Wr2, br2, mol_idx, mol_e);
    final_kernel<<<(kMols + 255) / 256, 256, 0, stream>>>(mol_e, (float*)d_out);
}
